// Attention_21672404975891
// MI455X (gfx1250) — matrix operand order repaired and checked by reading, not compiled, not run on hardware
//
#include <hip/hip_runtime.h>
#include <hip/hip_bf16.h>
#include <cstdint>
#include <cstddef>

// ---------------------------------------------------------------------------
// CDNA5 (gfx1250) attention pipeline.
//  - All matmuls: v_wmma_f32_16x16x32_f16, 128x128 block tile, 8 wave32 waves.
//  - Single uniform GEMM data path: A f16 [M,K], B f16 [N,K] (both ld=K), so
//    every staging load is a contiguous 16B copy -> CDNA5 async global->LDS
//    engine (GLOBAL_LOAD_ASYNC_TO_LDS_B128 + ASYNCcnt), double-buffered in
//    LDS to overlap the copies with WMMA.
//  - f32->f16 conversions + weight transposes hoisted into tiny memory-bound
//    prep kernels (~150MB @ 23.3TB/s ~ 7us, negligible vs ~137 GFLOP compute).
//  - Scores (4x2048x2048 f32 = 64MB) are materialized: they stay resident in
//    the 192MB L2, so scores->softmax->P@V round trips never touch HBM.
// ---------------------------------------------------------------------------

typedef _Float16 half_t;
typedef __attribute__((ext_vector_type(16))) _Float16 v16h;
typedef __attribute__((ext_vector_type(8)))  _Float16 v8h;
typedef __attribute__((ext_vector_type(4)))  _Float16 v4h;
typedef __attribute__((ext_vector_type(8)))  float    v8f;
typedef __attribute__((ext_vector_type(4)))  float    v4f;

#define BM 128
#define BN 128
#define BK 32
#define THREADS 256

// ---- CDNA5 async global->LDS probe (falls back to sync copies if absent) ----
#if defined(__gfx1250__) && \
    __has_builtin(__builtin_amdgcn_global_load_async_to_lds_b128) && \
    __has_builtin(__builtin_amdgcn_s_wait_asynccnt)
#define USE_ASYNC_LDS 1
#else
#define USE_ASYNC_LDS 0
#endif

// Exact parameter types per the compiler diagnostic:
//   arg0: int __vector(4) * in AS1 (printed '__device__'), non-const
//   arg1: int __vector(4) * in AS3 (LDS)
typedef int gv4i __attribute__((vector_size(16)));
typedef __attribute__((address_space(1))) gv4i* gptr_b128;
typedef __attribute__((address_space(3))) gv4i* lptr_b128;

// f32 -> f16 straight conversion (one vec4 per thread)
__global__ __launch_bounds__(THREADS)
void cvt_f16_kernel(const float* __restrict__ src, half_t* __restrict__ dst,
                    long long n4)
{
    long long i = (long long)blockIdx.x * THREADS + threadIdx.x;
    if (i < n4) {
        v4f v = *(const v4f*)(src + i * 4);
        v4h o;
        o[0] = (_Float16)v[0]; o[1] = (_Float16)v[1];
        o[2] = (_Float16)v[2]; o[3] = (_Float16)v[3];
        *(v4h*)(dst + i * 4) = o;
    }
}

// W f32 [K,N] -> Wt f16 [N,K] via 32x32 LDS tile transpose
__global__ __launch_bounds__(THREADS)
void cvt_tr_kernel(const float* __restrict__ W, half_t* __restrict__ Wt,
                   int K, int N)
{
    __shared__ float t[32][33];
    const int n0 = blockIdx.x * 32, k0 = blockIdx.y * 32;
    const int tx = threadIdx.x & 31, ty = threadIdx.x >> 5;
    #pragma unroll
    for (int i = 0; i < 4; ++i)
        t[ty + 8 * i][tx] = W[(size_t)(k0 + ty + 8 * i) * N + n0 + tx];
    __syncthreads();
    #pragma unroll
    for (int i = 0; i < 4; ++i)
        Wt[(size_t)(n0 + ty + 8 * i) * K + k0 + tx] = (half_t)t[tx][ty + 8 * i];
}

// Uniform WMMA GEMM: C = scale * (A[M,K] @ B[N,K]^T) (+ bias[N])
//   OUT_TRANS: store C transposed as [N][M] (used to produce vT directly)
template<bool HAS_BIAS, bool OUT_F16, bool OUT_TRANS>
__global__ __launch_bounds__(THREADS)
void gemm_wmma_kernel(const half_t* __restrict__ Ag, const half_t* __restrict__ Bg,
                      const float* __restrict__ bias, void* __restrict__ Cp,
                      int M, int N, int K,
                      long long strideA, long long strideB, long long strideC,
                      float scale)
{
    __shared__ __attribute__((aligned(32))) _Float16 As[2][BM * BK];  // [m][k]
    __shared__ __attribute__((aligned(32))) _Float16 Bs[2][BN * BK];  // [n][k]

    const int tid  = threadIdx.x;
    const int lane = tid & 31;
    const int wave = tid >> 5;
    const int l16  = lane & 15;
    const int hi   = (lane >> 4) & 1;
    const int waveM = wave & 3;    // 4 waves along M (32 rows each)
    const int waveN = wave >> 2;   // 2 waves along N (64 cols each)

    const int blockRow = blockIdx.y * BM;
    const int blockCol = blockIdx.x * BN;
    const long long batch = blockIdx.z;

    const half_t* A = Ag + batch * strideA;
    const half_t* B = Bg + batch * strideB;

    const v8f vzero = {0.f, 0.f, 0.f, 0.f, 0.f, 0.f, 0.f, 0.f};
    v8f acc[2][4];
    #pragma unroll
    for (int mi = 0; mi < 2; ++mi)
        #pragma unroll
        for (int nj = 0; nj < 4; ++nj)
            acc[mi][nj] = vzero;

    // Stage one 128x32 f16 tile pair into LDS buffer `buf`.
    // 2 x b128 per thread per matrix (512 x 16B = 8KB each).
    auto stage = [&](int buf, int kb) {
        #pragma unroll
        for (int it = 0; it < 2; ++it) {
            int idx = tid + it * THREADS;        // b128 id, 512 total
            int r   = idx >> 2;                  // 4 x b128 per 32-half row
            int c   = (idx & 3) << 3;
            const half_t* ga = A + (size_t)(blockRow + r) * K + kb + c;
            const half_t* gb = B + (size_t)(blockCol + r) * K + kb + c;
            _Float16* la = &As[buf][r * BK + c];
            _Float16* lb = &Bs[buf][r * BK + c];
#if USE_ASYNC_LDS
            __builtin_amdgcn_global_load_async_to_lds_b128(
                (gptr_b128)ga, (lptr_b128)la, 0, 0);
            __builtin_amdgcn_global_load_async_to_lds_b128(
                (gptr_b128)gb, (lptr_b128)lb, 0, 0);
#else
            *(v8h*)la = *(const v8h*)ga;
            *(v8h*)lb = *(const v8h*)gb;
#endif
        }
    };

    stage(0, 0);
    int buf = 0;
    for (int kb = 0; kb < K; kb += BK, buf ^= 1) {
        const int nxt = kb + BK;
        if (nxt < K) {
            stage(buf ^ 1, nxt);                 // prefetch next tile (DMA)
#if USE_ASYNC_LDS
            __builtin_amdgcn_s_wait_asynccnt(4); // wait only for current tile
#endif
        } else {
#if USE_ASYNC_LDS
            __builtin_amdgcn_s_wait_asynccnt(0);
#endif
        }
        __syncthreads();

        // Fragments per ISA 7.12.2 wave32 layouts.
        v16h afrag[2], bfrag[4];
        #pragma unroll
        for (int mi = 0; mi < 2; ++mi) {
            // lanes 0-15: row m, K 0..7 & 16..23 ; lanes 16-31: K 8..15 & 24..31
            const _Float16* ap = &As[buf][(waveM * 32 + mi * 16 + l16) * BK + hi * 8];
            v8h lo = *(const v8h*)ap;
            v8h hh = *(const v8h*)(ap + 16);
            afrag[mi] = __builtin_shufflevector(lo, hh,
                0, 1, 2, 3, 4, 5, 6, 7, 8, 9, 10, 11, 12, 13, 14, 15);
        }
        #pragma unroll
        for (int nj = 0; nj < 4; ++nj) {
            // lanes 0-15: col n, K 0..7 & 16..23 ; lanes 16-31: K 8..15 & 24..31
            const _Float16* bp = &Bs[buf][(waveN * 64 + nj * 16 + l16) * BK + hi * 8];
            v8h blo = *(const v8h*)bp;
            v8h bhh = *(const v8h*)(bp + 16);
            bfrag[nj] = __builtin_shufflevector(blo, bhh,
                0, 1, 2, 3, 4, 5, 6, 7, 8, 9, 10, 11, 12, 13, 14, 15);
        }
        #pragma unroll
        for (int mi = 0; mi < 2; ++mi)
            #pragma unroll
            for (int nj = 0; nj < 4; ++nj)
                acc[mi][nj] = __builtin_amdgcn_wmma_f32_16x16x32_f16(
                    false, afrag[mi], false, bfrag[nj],
                    (short)0, acc[mi][nj], false, false);

        __syncthreads();   // all waves done with `buf` before it is re-staged
    }

    // Epilogue: scale, bias, store (C layout: VGPR r -> row r / r+8, col = l16)
    #pragma unroll
    for (int mi = 0; mi < 2; ++mi) {
        #pragma unroll
        for (int nj = 0; nj < 4; ++nj) {
            int rowBase = blockRow + waveM * 32 + mi * 16 + hi * 8;
            int col     = blockCol + waveN * 64 + nj * 16 + l16;
            float bv = HAS_BIAS ? bias[col] : 0.0f;
            #pragma unroll
            for (int r = 0; r < 8; ++r) {
                float val = acc[mi][nj][r] * scale + bv;
                long long off = OUT_TRANS
                    ? batch * strideC + (long long)col * M + (rowBase + r)
                    : batch * strideC + (long long)(rowBase + r) * N + col;
                if (OUT_F16) ((half_t*)Cp)[off] = (half_t)val;
                else         ((float*)Cp)[off]  = val;
            }
        }
    }
}

// Row softmax with mask: one block per row of [B*S, S] scores.
__global__ __launch_bounds__(THREADS)
void softmax_kernel(const float* __restrict__ S, const int* __restrict__ mask,
                    half_t* __restrict__ P, int ncol, int s_rows)
{
    __shared__ float smax[8];
    __shared__ float ssum[8];
    const int row  = blockIdx.x;
    const int srow = row % s_rows;                 // mask is [1,S,S]
    const float* s = S + (size_t)row * ncol;
    const int*   m = mask + (size_t)srow * ncol;
    half_t*      p = P + (size_t)row * ncol;

    const int tid = threadIdx.x, lane = tid & 31, wv = tid >> 5;

    float lmax = -3.4e38f;
    for (int c = tid; c < ncol; c += THREADS) {
        float v = (m[c] == 0) ? -1e9f : s[c];
        lmax = fmaxf(lmax, v);
    }
    for (int o = 16; o > 0; o >>= 1) lmax = fmaxf(lmax, __shfl_xor(lmax, o, 32));
    if (lane == 0) smax[wv] = lmax;
    __syncthreads();
    if (tid == 0) {
        float t = smax[0];
        for (int i = 1; i < 8; ++i) t = fmaxf(t, smax[i]);
        smax[0] = t;
    }
    __syncthreads();
    const float rmax = smax[0];

    float lsum = 0.0f;
    for (int c = tid; c < ncol; c += THREADS) {
        float v = (m[c] == 0) ? -1e9f : s[c];
        lsum += __expf(v - rmax);
    }
    for (int o = 16; o > 0; o >>= 1) lsum += __shfl_xor(lsum, o, 32);
    if (lane == 0) ssum[wv] = lsum;
    __syncthreads();
    if (tid == 0) {
        float t = 0.0f;
        for (int i = 0; i < 8; ++i) t += ssum[i];
        ssum[0] = t;
    }
    __syncthreads();
    const float inv = 1.0f / ssum[0];

    for (int c = tid; c < ncol; c += THREADS) {
        float v = (m[c] == 0) ? -1e9f : s[c];
        p[c] = (half_t)(__expf(v - rmax) * inv);
    }
}

extern "C" void kernel_launch(void* const* d_in, const int* in_sizes, int n_in,
                              void* d_out, int out_size, void* d_ws, size_t ws_size,
                              hipStream_t stream) {
    (void)in_sizes; (void)n_in; (void)out_size; (void)ws_size;
    const int B = 4, S = 2048, D = 1024;
    const long long MS = (long long)B * S;          // 8192 rows

    const float* query = (const float*)d_in[0];
    const float* keyi  = (const float*)d_in[1];
    const float* value = (const float*)d_in[2];
    const int*   maskp = (const int*)  d_in[3];
    const float* Wq = (const float*)d_in[4];  const float* bq = (const float*)d_in[5];
    const float* Wk = (const float*)d_in[6];  const float* bk = (const float*)d_in[7];
    const float* Wv = (const float*)d_in[8];  const float* bv = (const float*)d_in[9];
    const float* Wo = (const float*)d_in[10]; const float* bo = (const float*)d_in[11];
    float* out = (float*)d_out;

    // Workspace layout (MB offsets). Scores round-trip stays in 192MB L2.
    char* ws = (char*)d_ws;
    const size_t MB = 1024ull * 1024ull;
    half_t* q16 = (half_t*)(ws +   0 * MB);   // f16(query)           16MB
    half_t* k16 = (half_t*)(ws +  16 * MB);   // f16(key)             16MB
    half_t* v16 = (half_t*)(ws +  32 * MB);   // f16(value)           16MB
    half_t* Wqt = (half_t*)(ws +  48 * MB);   // Wq^T f16              2MB
    half_t* Wkt = (half_t*)(ws +  50 * MB);
    half_t* Wvt = (half_t*)(ws +  52 * MB);
    half_t* Wot = (half_t*)(ws +  54 * MB);
    half_t* qf  = (half_t*)(ws +  56 * MB);   // q proj f16           16MB
    half_t* kf  = (half_t*)(ws +  72 * MB);   // k proj f16           16MB
    half_t* vT  = (half_t*)(ws +  88 * MB);   // v proj, [B][D][S]    16MB
    float*  sc  = (float*) (ws + 104 * MB);   // scores f32           64MB
    half_t* pf  = (half_t*)(ws + 168 * MB);   // P f16                32MB
    half_t* xf  = (half_t*)(ws +   0 * MB);   // x f16 (reuses q16)   16MB

    dim3 blk(THREADS);

    // ---- prep: activations f32->f16, weights f32[K,N] -> f16[N,K] ----
    {
        long long n4 = MS * D / 4;
        dim3 g((unsigned)((n4 + THREADS - 1) / THREADS));
        cvt_f16_kernel<<<g, blk, 0, stream>>>(query, q16, n4);
        cvt_f16_kernel<<<g, blk, 0, stream>>>(keyi,  k16, n4);
        cvt_f16_kernel<<<g, blk, 0, stream>>>(value, v16, n4);
        dim3 gt(D / 32, D / 32);
        cvt_tr_kernel<<<gt, blk, 0, stream>>>(Wq, Wqt, D, D);
        cvt_tr_kernel<<<gt, blk, 0, stream>>>(Wk, Wkt, D, D);
        cvt_tr_kernel<<<gt, blk, 0, stream>>>(Wv, Wvt, D, D);
        cvt_tr_kernel<<<gt, blk, 0, stream>>>(Wo, Wot, D, D);
    }

    // ---- q/k projections: [8192,1024] @ Wt[1024,1024]^T + bias -> f16 ----
    {
        dim3 g(D / BN, (unsigned)(MS / BM), 1);
        gemm_wmma_kernel<true, true, false><<<g, blk, 0, stream>>>(
            q16, Wqt, bq, qf, (int)MS, D, D, 0, 0, 0, 1.0f);
        gemm_wmma_kernel<true, true, false><<<g, blk, 0, stream>>>(
            k16, Wkt, bk, kf, (int)MS, D, D, 0, 0, 0, 1.0f);
    }
    // ---- v projection, stored transposed per batch: vT[b][d][t] ----
    {
        dim3 g(D / BN, S / BM, B);
        gemm_wmma_kernel<true, true, true><<<g, blk, 0, stream>>>(
            v16, Wvt, bv, vT, S, D, D,
            (long long)S * D, 0, (long long)D * S, 1.0f);
    }

    // ---- scores = q @ k^T / sqrt(D): both operands already [*,K] f16 ----
    {
        dim3 g(S / BN, S / BM, B);
        gemm_wmma_kernel<false, false, false><<<g, blk, 0, stream>>>(
            qf, kf, nullptr, sc, S, S, D,
            (long long)S * D, (long long)S * D, (long long)S * S, 0.03125f);
    }

    // ---- softmax rows (with mask) -> P f16 ----
    softmax_kernel<<<dim3((unsigned)MS), blk, 0, stream>>>(sc, maskp, pf, S, S);

    // ---- x = P @ v  (B operand is vT[d][t] = [N,K] f16) ----
    {
        dim3 g(D / BN, S / BM, B);
        gemm_wmma_kernel<false, true, false><<<g, blk, 0, stream>>>(
            pf, vT, nullptr, xf, S, D, S,
            (long long)S * S, (long long)D * S, (long long)S * D, 1.0f);
    }

    // ---- out = x @ Wo + bo -> f32 ----
    {
        dim3 g(D / BN, (unsigned)(MS / BM), 1);
        gemm_wmma_kernel<true, false, false><<<g, blk, 0, stream>>>(
            xf, Wot, bo, out, (int)MS, D, D, 0, 0, 0, 1.0f);
    }
}
